// GCN_LSTM_Peepholes_37890201486070
// MI455X (gfx1250) — matrix operand order repaired and checked by reading, not compiled, not run on hardware
//
#include <hip/hip_runtime.h>
#include <hip/hip_bf16.h>

#define NN    20000
#define EE    320000
#define INC   64
#define HID   512
#define G4    2048
#define NPRED 12
#define BN_EPS 1e-5f

typedef __attribute__((ext_vector_type(16))) __bf16 v16bf;
typedef __attribute__((ext_vector_type(8)))  float  v8f;

union BFrag { v16bf v; uint4 q[2]; };

__device__ __forceinline__ float sigm_(float x) { return 1.0f / (1.0f + __expf(-x)); }

// ---------------------------------------------------------------------------
// One GEMM pass: acc[2][4] += A(32xK) @ Wp(Kx64 packed), for one wave.
// Inlined with direct kernel-arg pointers so clang can infer global addrspace
// (global_load_b128 instead of flat_load_b128).
// A row-major [M,K] bf16. Wp packed in B-fragment order: tile (kt,nt) ->
// 512 contiguous bf16 laid out [lane][16 elems].
// ---------------------------------------------------------------------------
__device__ __forceinline__ void gemm_pass(
    const __hip_bfloat16* __restrict__ A, const __hip_bfloat16* __restrict__ Wp,
    v8f acc[2][4], int K, int tiles_n, int ntb, int m0, int lm, int hs, int lane)
{
    // A fragment rows: lane 0-15 -> M=lm with K 0..7 / 16..23,
    //                  lane 16-31 -> M=lm with K 8..15 / 24..31
    const __hip_bfloat16* Arow0 = A + (size_t)(m0 + lm) * K + (hs << 3);
    const __hip_bfloat16* Arow1 = Arow0 + (size_t)16 * K;
    const int ktiles = K >> 5;
    for (int kt = 0; kt < ktiles; ++kt) {
        BFrag a0, a1;
        a0.q[0] = *(const uint4*)(Arow0 + (kt << 5));
        a0.q[1] = *(const uint4*)(Arow0 + (kt << 5) + 16);
        a1.q[0] = *(const uint4*)(Arow1 + (kt << 5));
        a1.q[1] = *(const uint4*)(Arow1 + (kt << 5) + 16);
        __builtin_prefetch((const void*)(Arow0 + ((kt + 1) << 5)), 0, 3);
        __builtin_prefetch((const void*)(Arow1 + ((kt + 1) << 5)), 0, 3);
        const __hip_bfloat16* bt =
            Wp + (((size_t)(kt * tiles_n + ntb)) << 9) + (lane << 4);
#pragma unroll
        for (int t = 0; t < 4; ++t) {
            BFrag b;
            const uint4* bp = (const uint4*)(bt + ((size_t)t << 9));
            b.q[0] = bp[0];
            b.q[1] = bp[1];
            acc[0][t] = __builtin_amdgcn_wmma_f32_16x16x32_bf16(
                false, a0.v, false, b.v, (short)0, acc[0][t], false, false);
            acc[1][t] = __builtin_amdgcn_wmma_f32_16x16x32_bf16(
                false, a1.v, false, b.v, (short)0, acc[1][t], false, false);
        }
    }
}

// ---------------------------------------------------------------------------
// WMMA bf16 GEMM:  C[M, colrange] (+)= A1@W1 (+ A2@W2) (+ bias)
// Wave tile: 32 rows x 64 cols (8 accumulators, B fragment reused 2x).
// Block: 4 waves -> 32x256. M multiple of 32 (20000 = 625*32); Ncols mult 256.
// ---------------------------------------------------------------------------
__global__ __launch_bounds__(128) void wmma_gemm_bf16(
    const __hip_bfloat16* __restrict__ A1, const __hip_bfloat16* __restrict__ W1,
    const __hip_bfloat16* __restrict__ A2, const __hip_bfloat16* __restrict__ W2,
    const float* __restrict__ bias, float* __restrict__ C,
    int K, int Ncols, int ldc, int accum)
{
    const int lane = threadIdx.x & 31;
    const int wave = threadIdx.x >> 5;
    const int m0 = blockIdx.x << 5;
    const int n0 = (blockIdx.y << 8) + (wave << 6);
    const int tiles_n = Ncols >> 4;
    const int lm = lane & 15;
    const int hs = lane >> 4;          // lane-half select
    const int ntb = n0 >> 4;

    v8f zero = {0.f,0.f,0.f,0.f,0.f,0.f,0.f,0.f};
    v8f acc[2][4] = {{zero, zero, zero, zero}, {zero, zero, zero, zero}};

    gemm_pass(A1, W1, acc, K, tiles_n, ntb, m0, lm, hs, lane);
    if (A2 != nullptr)
        gemm_pass(A2, W2, acc, K, tiles_n, ntb, m0, lm, hs, lane);

    // C/D layout: VGPR r -> row m0 + s*16 + (hs?8:0) + r, col n0 + t*16 + lm
#pragma unroll
    for (int s = 0; s < 2; ++s) {
        float* crow = C + (size_t)(m0 + (s << 4) + (hs << 3)) * ldc;
#pragma unroll
        for (int t = 0; t < 4; ++t) {
            int col = n0 + (t << 4) + lm;
            float badd = bias ? bias[col] : 0.0f;
            float* cp = crow + col;
#pragma unroll
            for (int r = 0; r < 8; ++r) {
                float v = acc[s][t][r] + badd;
                if (accum) v += cp[(size_t)r * ldc];
                cp[(size_t)r * ldc] = v;
            }
        }
    }
}

// Pack f32 weights [K, ldW] (tiles*16 cols from colStart+16*ntOff) into bf16
// B-fragment layout. tilesTot = total column tiles of destination matrix,
// ntOff = column-tile offset inside destination (for concatenated packs).
__global__ void pack_b(const float* __restrict__ W, int ldW, int colStart, int K,
                       int tiles, int tilesTot, int ntOff,
                       __hip_bfloat16* __restrict__ out)
{
    int id = blockIdx.x * 256 + threadIdx.x;
    int total = (K >> 5) * tiles * 512;
    if (id >= total) return;
    int within = id & 511;
    int tile = id >> 9;
    int kt = tile / tiles;
    int ntl = tile - kt * tiles;
    int lane = within >> 4;
    int e = within & 15;
    int k = (kt << 5) + ((e >> 3) << 4) + ((lane >> 4) << 3) + (e & 7);  // lanes 16-31: K+8
    int col = colStart + ((ntOff + ntl) << 4) + (lane & 15);
    out[(((size_t)(kt * tilesTot + ntOff + ntl)) << 9) + within] =
        __float2bfloat16(W[(size_t)k * ldW + col]);
}

__global__ void f32_to_bf16_k(const float* __restrict__ in,
                              __hip_bfloat16* __restrict__ out, int n)
{
    int id = blockIdx.x * 256 + threadIdx.x;
    if (id < n) out[id] = __float2bfloat16(in[id]);
}

__global__ void zero_f32(float* __restrict__ p, int n)
{
    int id = blockIdx.x * 256 + threadIdx.x;
    if (id < n) p[id] = 0.0f;
}

// agg[dst[e]] += ew[e] * H[src[e]] : 128 threads/edge, float4 per thread
__global__ void gcn_scatter(const float* __restrict__ H,
                            const int* __restrict__ srcI,
                            const int* __restrict__ dstI,
                            const float* __restrict__ ew,
                            float* __restrict__ agg)
{
    int id = blockIdx.x * 256 + threadIdx.x;
    int e = id >> 7;
    if (e >= EE) return;
    int c = (id & 127) << 2;
    float w = ew[e];
    const float4 v = *(const float4*)(H + (size_t)srcI[e] * HID + c);
    float* dp = agg + (size_t)dstI[e] * HID + c;
    unsafeAtomicAdd(dp + 0, w * v.x);
    unsafeAtomicAdd(dp + 1, w * v.y);
    unsafeAtomicAdd(dp + 2, w * v.z);
    unsafeAtomicAdd(dp + 3, w * v.w);
}

// Per-feature batch stats -> scale/shift (bias cancels under train-mode BN)
__global__ void bn_stats(const float* __restrict__ X,
                         const float* __restrict__ gamma,
                         const float* __restrict__ beta,
                         float* __restrict__ scale, float* __restrict__ shift, int n)
{
    __shared__ float s1[256], s2[256];
    int f = blockIdx.x, t = threadIdx.x;
    float sum = 0.f, sq = 0.f;
    for (int r = t; r < n; r += 256) {
        float v = X[(size_t)r * HID + f];
        sum += v; sq += v * v;
    }
    s1[t] = sum; s2[t] = sq;
    __syncthreads();
    for (int s = 128; s > 0; s >>= 1) {
        if (t < s) { s1[t] += s1[t + s]; s2[t] += s2[t + s]; }
        __syncthreads();
    }
    if (t == 0) {
        float mu = s1[0] / (float)n;
        float var = s2[0] / (float)n - mu * mu;
        float rs = rsqrtf(var + BN_EPS);
        float sc = gamma[f] * rs;
        scale[f] = sc;
        shift[f] = beta[f] - mu * sc;
    }
}

__global__ void bn_relu_bf16(const float* __restrict__ X,
                             const float* __restrict__ scale,
                             const float* __restrict__ shift,
                             __hip_bfloat16* __restrict__ out, int n)
{
    int id = blockIdx.x * 256 + threadIdx.x;
    if (id >= n) return;
    int f = id & (HID - 1);
    float y = fmaf(X[id], scale[f], shift[f]);
    out[id] = __float2bfloat16(fmaxf(y, 0.0f));
}

// cy = sigmoid(f)*c + sigmoid(i)*tanh(g)   (f/g slices already hold peepholes)
__global__ void lstm_ew1(const float* __restrict__ G, float* __restrict__ Cst,
                         __hip_bfloat16* __restrict__ cb, int hasC, int n)
{
    int id = blockIdx.x * 256 + threadIdx.x;
    if (id >= n) return;
    int row = id >> 9, j = id & (HID - 1);
    const float* g4 = G + ((size_t)row << 11);
    float ig = sigm_(g4[j]);
    float gg = tanhf(g4[2 * HID + j]);
    float cy = ig * gg;
    if (hasC) cy += sigm_(g4[HID + j]) * Cst[id];
    Cst[id] = cy;
    cb[id] = __float2bfloat16(cy);
}

// hy = sigmoid(o + cy@Wch_o)*tanh(cy); o-peephole was written into g-slice
__global__ void lstm_ew2(const float* __restrict__ G, const float* __restrict__ Cst,
                         float* __restrict__ Hst, __hip_bfloat16* __restrict__ hb, int n)
{
    int id = blockIdx.x * 256 + threadIdx.x;
    if (id >= n) return;
    int row = id >> 9, j = id & (HID - 1);
    const float* g4 = G + ((size_t)row << 11);
    float o = sigm_(g4[3 * HID + j] + g4[2 * HID + j]);
    float hy = o * tanhf(Cst[id]);
    Hst[id] = hy;
    hb[id] = __float2bfloat16(hy);
}

__global__ void out_proj(const float* __restrict__ H, const float* __restrict__ Wo,
                         const float* __restrict__ bo, float* __restrict__ out, int n)
{
    int id = blockIdx.x * 256 + threadIdx.x;
    if (id >= n) return;
    int row = id / NPRED, p = id - row * NPRED;
    const float* h = H + ((size_t)row << 9);
    float acc = bo[p];
    for (int k = 0; k < HID; ++k) acc = fmaf(h[k], Wo[k * NPRED + p], acc);
    out[id] = acc;
}

// ---------------------------------------------------------------------------
extern "C" void kernel_launch(void* const* d_in, const int* in_sizes, int n_in,
                              void* d_out, int out_size, void* d_ws, size_t ws_size,
                              hipStream_t stream)
{
    const float* x      = (const float*)d_in[0];
    const int*   eidx   = (const int*)d_in[1];
    const float* ew     = (const float*)d_in[2];
    const float* gcn0_W = (const float*)d_in[3];
    const float* gcn_W  = (const float*)d_in[5];
    const float* bn_g   = (const float*)d_in[7];
    const float* bn_b   = (const float*)d_in[8];
    const float* w_ih   = (const float*)d_in[9];
    const float* w_hh   = (const float*)d_in[10];
    const float* w_ch   = (const float*)d_in[11];
    const float* lstm_b = (const float*)d_in[12];
    const float* out_W  = (const float*)d_in[13];
    const float* out_b  = (const float*)d_in[14];
    float* out = (float*)d_out;
    (void)in_sizes; (void)n_in; (void)out_size; (void)ws_size;

    char* base = (char*)d_ws;
    size_t off = 0;
    auto alloc = [&](size_t bytes) -> void* {
        void* p = base + off;
        off += (bytes + 255) & ~((size_t)255);
        return p;
    };

    float* G   = (float*)alloc((size_t)NN * G4 * 4);   // gates / GCN gemm out
    float* AGG = (float*)alloc((size_t)NN * HID * 4);  // GCN agg, reused as LSTM c
    float* Hf  = (float*)alloc((size_t)NN * HID * 4);
    __hip_bfloat16* act = (__hip_bfloat16*)alloc((size_t)NN * HID * 2);
    __hip_bfloat16* xb  = (__hip_bfloat16*)alloc((size_t)NN * INC * 2);
    __hip_bfloat16* cb  = (__hip_bfloat16*)alloc((size_t)NN * HID * 2);
    __hip_bfloat16* hb  = (__hip_bfloat16*)alloc((size_t)NN * HID * 2);
    float* scale = (float*)alloc(HID * 4);
    float* shift = (float*)alloc(HID * 4);
    __hip_bfloat16* Wg0p   = (__hip_bfloat16*)alloc((size_t)INC * HID * 2);
    __hip_bfloat16* Wg1p   = (__hip_bfloat16*)alloc((size_t)HID * HID * 2);
    __hip_bfloat16* Wg2p   = (__hip_bfloat16*)alloc((size_t)HID * HID * 2);
    __hip_bfloat16* Wih0p  = (__hip_bfloat16*)alloc((size_t)HID * G4 * 2);
    __hip_bfloat16* Wih1p  = (__hip_bfloat16*)alloc((size_t)HID * G4 * 2);
    __hip_bfloat16* Whh1p  = (__hip_bfloat16*)alloc((size_t)HID * G4 * 2);
    __hip_bfloat16* Wc0op  = (__hip_bfloat16*)alloc((size_t)HID * HID * 2);
    __hip_bfloat16* Wc1fgp = (__hip_bfloat16*)alloc((size_t)HID * 1024 * 2);
    __hip_bfloat16* Wc1op  = (__hip_bfloat16*)alloc((size_t)HID * HID * 2);

    const int* srcI = eidx;
    const int* dstI = eidx + EE;

    // --- input to bf16, pack all weights into WMMA B-fragment layout -------
    { int tot = NN * INC; f32_to_bf16_k<<<(tot + 255) / 256, 256, 0, stream>>>(x, xb, tot); }
    { int tot = (INC >> 5) * 32 * 512;
      pack_b<<<(tot + 255) / 256, 256, 0, stream>>>(gcn0_W, HID, 0, INC, 32, 32, 0, Wg0p); }
    { int tot = (HID >> 5) * 32 * 512;
      pack_b<<<(tot + 255) / 256, 256, 0, stream>>>(gcn_W,             HID, 0, HID, 32, 32, 0, Wg1p);
      pack_b<<<(tot + 255) / 256, 256, 0, stream>>>(gcn_W + HID * HID, HID, 0, HID, 32, 32, 0, Wg2p);
      pack_b<<<(tot + 255) / 256, 256, 0, stream>>>(w_ch,              1536, 1024, HID, 32, 32, 0, Wc0op);
      pack_b<<<(tot + 255) / 256, 256, 0, stream>>>(w_ch + HID * 1536, 1536, 1024, HID, 32, 32, 0, Wc1op);
      // packed [f-peep | g-peep] -> accumulates into gates cols [H, 3H)
      pack_b<<<(tot + 255) / 256, 256, 0, stream>>>(w_ch + HID * 1536, 1536, 512, HID, 32, 64, 0,  Wc1fgp);
      pack_b<<<(tot + 255) / 256, 256, 0, stream>>>(w_ch + HID * 1536, 1536, 0,   HID, 32, 64, 32, Wc1fgp); }
    { int tot = (HID >> 5) * 128 * 512;
      pack_b<<<(tot + 255) / 256, 256, 0, stream>>>(w_ih,            G4, 0, HID, 128, 128, 0, Wih0p);
      pack_b<<<(tot + 255) / 256, 256, 0, stream>>>(w_ih + HID * G4, G4, 0, HID, 128, 128, 0, Wih1p);
      pack_b<<<(tot + 255) / 256, 256, 0, stream>>>(w_hh + HID * G4, G4, 0, HID, 128, 128, 0, Whh1p); }

    // --- GCN stack (bias dropped: no-op under train-mode BN) ---------------
    for (int l = 0; l < 3; ++l) {
        const __hip_bfloat16* Ain = (l == 0) ? xb : act;
        int K = (l == 0) ? INC : HID;
        const __hip_bfloat16* Wp = (l == 0) ? Wg0p : (l == 1 ? Wg1p : Wg2p);
        wmma_gemm_bf16<<<dim3(NN / 32, HID / 256), 128, 0, stream>>>(
            Ain, Wp, nullptr, nullptr, nullptr, G, K, HID, HID, 0);
        { int tot = NN * HID; zero_f32<<<tot / 256, 256, 0, stream>>>(AGG, tot); }
        gcn_scatter<<<(EE * 128) / 256, 256, 0, stream>>>(G, srcI, dstI, ew, AGG);
        bn_stats<<<HID, 256, 0, stream>>>(AGG, bn_g + l * HID, bn_b + l * HID, scale, shift, NN);
        { int tot = NN * HID; bn_relu_bf16<<<tot / 256, 256, 0, stream>>>(AGG, scale, shift, act, tot); }
    }

    float* Cst = AGG;  // cell state reuses AGG

    // --- LSTM layer 0 (h=c=0: skip Whh and c-peepholes) --------------------
    wmma_gemm_bf16<<<dim3(NN / 32, G4 / 256), 128, 0, stream>>>(
        act, Wih0p, nullptr, nullptr, lstm_b, G, HID, G4, G4, 0);
    { int tot = NN * HID; lstm_ew1<<<tot / 256, 256, 0, stream>>>(G, Cst, cb, 0, tot); }
    wmma_gemm_bf16<<<dim3(NN / 32, HID / 256), 128, 0, stream>>>(
        cb, Wc0op, nullptr, nullptr, nullptr, G + 2 * HID, HID, HID, G4, 0);  // o-peep -> g-slice
    { int tot = NN * HID; lstm_ew2<<<tot / 256, 256, 0, stream>>>(G, Cst, Hf, hb, tot); }

    // --- LSTM layer 1 ------------------------------------------------------
    wmma_gemm_bf16<<<dim3(NN / 32, G4 / 256), 128, 0, stream>>>(
        act, Wih1p, hb, Whh1p, lstm_b + G4, G, HID, G4, G4, 0);
    wmma_gemm_bf16<<<dim3(NN / 32, 1024 / 256), 128, 0, stream>>>(
        cb, Wc1fgp, nullptr, nullptr, nullptr, G + HID, HID, 1024, G4, 1);    // c-peep += f,g
    { int tot = NN * HID; lstm_ew1<<<tot / 256, 256, 0, stream>>>(G, Cst, cb, 1, tot); }
    wmma_gemm_bf16<<<dim3(NN / 32, HID / 256), 128, 0, stream>>>(
        cb, Wc1op, nullptr, nullptr, nullptr, G + 2 * HID, HID, HID, G4, 0);  // o-peep -> g-slice
    { int tot = NN * HID; lstm_ew2<<<tot / 256, 256, 0, stream>>>(G, Cst, Hf, hb, tot); }

    // --- output projection -------------------------------------------------
    { int tot = NN * NPRED;
      out_proj<<<(tot + 255) / 256, 256, 0, stream>>>(Hf, out_W, out_b, out, tot); }
}
